// UnfusedVisionAttention_56521769615495
// MI455X (gfx1250) — matrix operand order repaired and checked by reading, not compiled, not run on hardware
//
#include <hip/hip_runtime.h>

typedef __attribute__((ext_vector_type(16))) __bf16 v16bf;
typedef __attribute__((ext_vector_type(8)))  float  v8f;

#define SEQ       16384
#define HIDDEN    1280
#define NHEADS    16
#define HDIM      80
#define WIN       64
#define NWIN      256
#define QKV_COLS  3840
#define LDSROW    72          // LDS row stride in bf16 elems (144B, 16B-aligned)

// ---------- bf16 helpers (RNE) ----------
__device__ __forceinline__ unsigned short f2bf(float f) {
    unsigned int u = __float_as_uint(f);
    u += 0x7FFFu + ((u >> 16) & 1u);
    return (unsigned short)(u >> 16);
}
__device__ __forceinline__ float bf2f(unsigned short h) {
    return __uint_as_float(((unsigned int)h) << 16);
}

// ---------- CDNA5 async global->LDS copy (ASYNCcnt-tracked) ----------
__device__ __forceinline__ void async_b128(unsigned lds_off, unsigned gl_off,
                                           const void* sbase) {
    asm volatile("global_load_async_to_lds_b128 %0, %1, %2"
                 :: "v"(lds_off), "v"(gl_off), "s"(sbase) : "memory");
}
__device__ __forceinline__ unsigned lds_addr(const void* p) {
    return (unsigned)(size_t)p;   // low 32 bits of generic ptr == LDS byte offset
}

// ---------- WMMA fragment loaders (CDNA5 16-bit layouts, wave32) ----------
// A (16x32, MxK): lane m=lane&15, half=lane>>4; elems 0..7 -> K=8h+e, 8..15 -> K=16+8h+(e-8)
__device__ __forceinline__ v16bf load_frag_a(const unsigned short* lds, int stride,
                                             int row0, int k0, int lane) {
    const int m = lane & 15, h = lane >> 4;
    const unsigned short* p = lds + (row0 + m) * stride + k0;
    v16bf a;
    ((uint4*)&a)[0] = *(const uint4*)(p + 8 * h);
    ((uint4*)&a)[1] = *(const uint4*)(p + 16 + 8 * h);
    return a;
}
// B (32x16, KxN) from K-contiguous transposed tile Bt[n][k]: lane n=lane&15; elems 0..7 -> K=8h+e, 8..15 -> K=16+8h+(e-8)
__device__ __forceinline__ v16bf load_frag_b(const unsigned short* lds, int stride,
                                             int col0, int k0, int lane) {
    const int n = lane & 15, h = lane >> 4;
    const unsigned short* p = lds + (col0 + n) * stride + k0;
    v16bf b;
    ((uint4*)&b)[0] = *(const uint4*)(p + 8 * h);
    ((uint4*)&b)[1] = *(const uint4*)(p + 16 + 8 * h);
    return b;
}

// =====================================================================
// Kernel 0: bulk fp32 -> bf16 (vectorized, n multiple of 1024)
// =====================================================================
__global__ __launch_bounds__(256) void cvt_bf16_kernel(
    const float* __restrict__ src, unsigned short* __restrict__ dst) {
    const size_t i = ((size_t)blockIdx.x * 256 + threadIdx.x) * 4;
    const float4 v = *(const float4*)(src + i);
    uint2 pk;
    pk.x = (unsigned)f2bf(v.x) | ((unsigned)f2bf(v.y) << 16);
    pk.y = (unsigned)f2bf(v.z) | ((unsigned)f2bf(v.w) << 16);
    *(uint2*)(dst + i) = pk;
}

// =====================================================================
// Unified bf16 GEMM: C = A @ B^T + bias, A[M][K], B[N][K] both bf16.
// 256 thr = 8 waves; block tile 128x128; wave tile 64x32; K-step 64;
// double-buffered LDS fed by global_load_async_to_lds_b128.
// =====================================================================
template <bool BF16_OUT>
__global__ __launch_bounds__(256) void gemm_bf16_async_kernel(
    const unsigned short* __restrict__ A, const unsigned short* __restrict__ B,
    const float* __restrict__ bias, void* __restrict__ Cout,
    const int K, const int N) {
    __shared__ unsigned short As[2][128 * LDSROW];
    __shared__ unsigned short Bs[2][128 * LDSROW];

    const int tid = threadIdx.x;
    const int lane = tid & 31, wid = tid >> 5;
    const int m0 = blockIdx.y * 128, n0 = blockIdx.x * 128;
    const int waveM = wid & 1, waveN = wid >> 1;

    const unsigned asBase = lds_addr(&As[0][0]);
    const unsigned bsBase = lds_addr(&Bs[0][0]);
    const unsigned stageB = 128 * LDSROW * 2;

    const int r0 = tid >> 3;       // 0..31
    const int seg = tid & 7;       // 16B chunk within 128B row

    // issue one 128x64 bf16 tile of A and of B into stage (k&1): 8 async instr/wave
    auto issueStage = [&](int kidx) {
        const int st = kidx & 1;
        const int kk = kidx * 64;
#pragma unroll
        for (int p = 0; p < 4; ++p) {
            const int row = p * 32 + r0;
            const unsigned ld = (unsigned)(st * (int)stageB + row * (LDSROW * 2) + seg * 16);
            const unsigned va = (unsigned)(((size_t)(m0 + row) * K + kk) * 2) + seg * 16;
            async_b128(asBase + ld, va, A);
            const unsigned vb = (unsigned)(((size_t)(n0 + row) * K + kk) * 2) + seg * 16;
            async_b128(bsBase + ld, vb, B);
        }
    };

    const v8f zero = {0.f, 0.f, 0.f, 0.f, 0.f, 0.f, 0.f, 0.f};
    v8f acc[4][2];
#pragma unroll
    for (int mt = 0; mt < 4; ++mt)
#pragma unroll
        for (int nt = 0; nt < 2; ++nt) acc[mt][nt] = zero;

    const int NK = K / 64;
    issueStage(0);
    for (int k = 0; k < NK; ++k) {
        const int st = k & 1;
        if (k + 1 < NK) {
            issueStage(k + 1);
            asm volatile("s_wait_asynccnt 0x8" ::: "memory");  // stage k landed
        } else {
            asm volatile("s_wait_asynccnt 0x0" ::: "memory");
        }
        __syncthreads();
        const unsigned short* pa = &As[st][0];
        const unsigned short* pb = &Bs[st][0];
#pragma unroll
        for (int ks = 0; ks < 2; ++ks) {
            v16bf af[4], bfg[2];
#pragma unroll
            for (int mt = 0; mt < 4; ++mt)
                af[mt] = load_frag_a(pa, LDSROW, 64 * waveM + 16 * mt, ks * 32, lane);
#pragma unroll
            for (int nt = 0; nt < 2; ++nt)
                bfg[nt] = load_frag_b(pb, LDSROW, 32 * waveN + 16 * nt, ks * 32, lane);
#pragma unroll
            for (int mt = 0; mt < 4; ++mt)
#pragma unroll
                for (int nt = 0; nt < 2; ++nt)
                    acc[mt][nt] = __builtin_amdgcn_wmma_f32_16x16x32_bf16(
                        false, af[mt], false, bfg[nt], (short)0, acc[mt][nt], false, false);
        }
        __syncthreads();   // stage k fully consumed before its buffer is reused
    }

    const int half = lane >> 4, ln = lane & 15;
#pragma unroll
    for (int mt = 0; mt < 4; ++mt)
#pragma unroll
        for (int nt = 0; nt < 2; ++nt) {
            const int col = n0 + 32 * waveN + 16 * nt + ln;
            const float bv = bias[col];
#pragma unroll
            for (int j = 0; j < 8; ++j) {
                const int row = m0 + 64 * waveM + 16 * mt + j + 8 * half;
                const float v = acc[mt][nt][j] + bv;
                if (BF16_OUT)
                    ((unsigned short*)Cout)[(size_t)row * N + col] = f2bf(v);
                else
                    ((float*)Cout)[(size_t)row * N + col] = v;
            }
        }
}

// =====================================================================
// Kernel 2: in-place 2D RoPE on q and k slices of bf16 qkv
// =====================================================================
__global__ __launch_bounds__(256) void rope_kernel(
    unsigned short* __restrict__ qkv,
    const float* __restrict__ cosT, const float* __restrict__ sinT) {
    const int idx = blockIdx.x * 256 + threadIdx.x;   // SEQ*NHEADS*40 total
    const int d = idx % 40;
    const int t = idx / 40;
    const int hh = t % NHEADS;
    const int s = t / NHEADS;
    const float c = cosT[s * 40 + d];
    const float sn = sinT[s * 40 + d];

    unsigned short* q = qkv + (size_t)s * QKV_COLS + hh * HDIM;
    float t1 = bf2f(q[d]), t2 = bf2f(q[d + 40]);
    q[d]      = f2bf(t1 * c - t2 * sn);
    q[d + 40] = f2bf(t2 * c + t1 * sn);

    unsigned short* k = qkv + (size_t)s * QKV_COLS + HIDDEN + hh * HDIM;
    t1 = bf2f(k[d]); t2 = bf2f(k[d + 40]);
    k[d]      = f2bf(t1 * c - t2 * sn);
    k[d + 40] = f2bf(t2 * c + t1 * sn);
}

// =====================================================================
// Kernel 3: windowed attention per (window, head); 128 thr = 4 waves
// Q/K tiles via async b128; scores 64x64 (K=80 pad 96), softmax, PV 64x80
// =====================================================================
__global__ __launch_bounds__(128) void attn_kernel(
    const unsigned short* __restrict__ qkv, unsigned short* __restrict__ out) {
    __shared__ unsigned short Qs[64 * 96];
    __shared__ unsigned short Ks[64 * 96];
    __shared__ unsigned short Vt[80 * 72];   // transposed: Vt[d][s]
    __shared__ unsigned short Ps[64 * 72];

    const int w = blockIdx.x, h = blockIdx.y;
    const int s0 = w * WIN;
    const int tid = threadIdx.x, lane = tid & 31, wid = tid >> 5;

    // Q and K rows (80 bf16 = 160B = 10 x 16B chunks per row): async to LDS
    const unsigned qsBase = lds_addr(Qs);
    const unsigned ksBase = lds_addr(Ks);
#pragma unroll
    for (int i = 0; i < 5; ++i) {
        const int chunk = tid + 128 * i;          // 0..639
        const int r = chunk / 10, sg = chunk % 10;
        const unsigned lq = qsBase + (unsigned)(r * 192 + sg * 16);
        const unsigned vq = (unsigned)(((size_t)(s0 + r) * QKV_COLS + h * HDIM) * 2) + sg * 16;
        async_b128(lq, vq, qkv);
        const unsigned lk = ksBase + (unsigned)(r * 192 + sg * 16);
        const unsigned vk = (unsigned)(((size_t)(s0 + r) * QKV_COLS + HIDDEN + h * HDIM) * 2) + sg * 16;
        async_b128(lk, vk, qkv);
    }
    {   // zero-pad cols 80..95 of Q/K
        const int r = tid & 63;
        unsigned short* dst = ((tid < 64) ? Qs : Ks) + r * 96;
        const uint4 z = make_uint4(0u, 0u, 0u, 0u);
        *(uint4*)(dst + 80) = z;
        *(uint4*)(dst + 88) = z;
    }
    // V transposed into LDS
    for (int e = tid; e < HDIM * WIN; e += 128) {
        const int d = e >> 6, s = e & 63;
        Vt[d * 72 + s] = qkv[(size_t)(s0 + s) * QKV_COLS + 2 * HIDDEN + h * HDIM + d];
    }
    asm volatile("s_wait_asynccnt 0x0" ::: "memory");
    __syncthreads();

    // ---- scores = Q K^T : wave handles rows [16*wid, +16), all 64 cols
    const v8f zero = {0.f, 0.f, 0.f, 0.f, 0.f, 0.f, 0.f, 0.f};
    v8f sc[4];
#pragma unroll
    for (int nt = 0; nt < 4; ++nt) sc[nt] = zero;
#pragma unroll
    for (int ks = 0; ks < 3; ++ks) {
        const int k0 = ks * 32;
        const v16bf af = load_frag_a(Qs, 96, 16 * wid, k0, lane);
#pragma unroll
        for (int nt = 0; nt < 4; ++nt) {
            const v16bf bfg = load_frag_b(Ks, 96, 16 * nt, k0, lane);
            sc[nt] = __builtin_amdgcn_wmma_f32_16x16x32_bf16(
                false, af, false, bfg, (short)0, sc[nt], false, false);
        }
    }

    // ---- softmax per row (lane = col, 8 rows/lane, 16-lane halves)
    const float scale = 0.11180339887498949f;   // 1/sqrt(80)
    const int half = lane >> 4, ln = lane & 15;
#pragma unroll
    for (int j = 0; j < 8; ++j) {
        float m = sc[0][j];
#pragma unroll
        for (int nt = 1; nt < 4; ++nt) m = fmaxf(m, sc[nt][j]);
#pragma unroll
        for (int off = 1; off < 16; off <<= 1) m = fmaxf(m, __shfl_xor(m, off, 32));
        float sum = 0.f;
#pragma unroll
        for (int nt = 0; nt < 4; ++nt) {
            const float p = __expf((sc[nt][j] - m) * scale);
            sc[nt][j] = p;
            sum += p;
        }
#pragma unroll
        for (int off = 1; off < 16; off <<= 1) sum += __shfl_xor(sum, off, 32);
        const float inv = 1.0f / sum;
        const int row = 16 * wid + j + 8 * half;
#pragma unroll
        for (int nt = 0; nt < 4; ++nt)
            Ps[row * 72 + 16 * nt + ln] = f2bf(sc[nt][j] * inv);
    }
    __syncthreads();

    // ---- out = P V : wave rows [16*wid, +16), cols 80 (5 N-tiles), K=64
    v8f oc[5];
#pragma unroll
    for (int nt = 0; nt < 5; ++nt) oc[nt] = zero;
#pragma unroll
    for (int ks = 0; ks < 2; ++ks) {
        const int k0 = ks * 32;
        const v16bf af = load_frag_a(Ps, 72, 16 * wid, k0, lane);
#pragma unroll
        for (int nt = 0; nt < 5; ++nt) {
            const v16bf bfg = load_frag_b(Vt, 72, 16 * nt, k0, lane);
            oc[nt] = __builtin_amdgcn_wmma_f32_16x16x32_bf16(
                false, af, false, bfg, (short)0, oc[nt], false, false);
        }
    }
#pragma unroll
    for (int nt = 0; nt < 5; ++nt)
#pragma unroll
        for (int j = 0; j < 8; ++j) {
            const int row = s0 + 16 * wid + j + 8 * half;
            out[(size_t)row * HIDDEN + h * HDIM + 16 * nt + ln] = f2bf(oc[nt][j]);
        }
}

// =====================================================================
extern "C" void kernel_launch(void* const* d_in, const int* in_sizes, int n_in,
                              void* d_out, int out_size, void* d_ws, size_t ws_size,
                              hipStream_t stream) {
    const float* x      = (const float*)d_in[0];
    const float* cosT   = (const float*)d_in[1];
    const float* sinT   = (const float*)d_in[2];
    const float* qkv_w  = (const float*)d_in[3];
    const float* qkv_b  = (const float*)d_in[4];
    const float* proj_w = (const float*)d_in[5];
    const float* proj_b = (const float*)d_in[6];
    // d_in[7] = cu_seqlens (implied by WIN/NWIN constants)

    // workspace layout (bf16 elems). attn output aliases xb (xb dead after gemm1).
    unsigned short* xb      = (unsigned short*)d_ws;                       // [SEQ][1280]
    unsigned short* attn_ws = xb;                                          // alias
    unsigned short* wb      = xb + (size_t)SEQ * HIDDEN;                   // [3840][1280]
    unsigned short* pb      = wb + (size_t)QKV_COLS * HIDDEN;              // [1280][1280]
    unsigned short* qkv_ws  = pb + (size_t)HIDDEN * HIDDEN;                // [SEQ][3840]
    float* out = (float*)d_out;

    // fp32 -> bf16 pre-conversion (operands then live bf16 in L2 for the GEMMs)
    cvt_bf16_kernel<<<dim3((SEQ * HIDDEN) / 1024), dim3(256), 0, stream>>>(x, xb);
    cvt_bf16_kernel<<<dim3((QKV_COLS * HIDDEN) / 1024), dim3(256), 0, stream>>>(qkv_w, wb);
    cvt_bf16_kernel<<<dim3((HIDDEN * HIDDEN) / 1024), dim3(256), 0, stream>>>(proj_w, pb);

    gemm_bf16_async_kernel<true><<<dim3(QKV_COLS / 128, SEQ / 128), dim3(256), 0, stream>>>(
        xb, wb, qkv_b, (void*)qkv_ws, HIDDEN, QKV_COLS);

    rope_kernel<<<dim3((SEQ * NHEADS * 40) / 256), dim3(256), 0, stream>>>(
        qkv_ws, cosT, sinT);

    attn_kernel<<<dim3(NWIN, NHEADS), dim3(128), 0, stream>>>(qkv_ws, attn_ws);

    gemm_bf16_async_kernel<false><<<dim3(HIDDEN / 128, SEQ / 128), dim3(256), 0, stream>>>(
        attn_ws, pb, proj_b, (void*)out, HIDDEN, HIDDEN);
}
